// LDEB_64235530879110
// MI455X (gfx1250) — matrix operand order repaired and checked by reading, not compiled, not run on hardware
//
#include <hip/hip_runtime.h>

typedef __attribute__((ext_vector_type(16))) _Float16 v16h;
typedef __attribute__((ext_vector_type(8)))  _Float16 v8h;
typedef __attribute__((ext_vector_type(2)))  __fp16   fp16x2;
typedef __attribute__((ext_vector_type(8)))  float    v8f;
typedef __attribute__((ext_vector_type(4)))  float    f32x4;
typedef __attribute__((ext_vector_type(8)))  unsigned int u32x8;

#define IMG  512
#define SLAB 4      // slab origin offset: x0-4 keeps fast-path loads 16B aligned
#define TPW  8      // tiles per wave
#define WVS  8      // waves per block
#define WPAD 40     // LDS row padding (halves): 80B stride, 16B aligned, 2-way max conflict

// Zero-padded pixel load (TF-SAME zero padding); must return finite values
// everywhere because WMMA propagates NaN and 0*Inf -> NaN.
__device__ __forceinline__ float loadpix(const float* __restrict__ x, int b, int r, int c) {
    bool ok = (r >= 0) & (r < IMG) & (c >= 0) & (c < IMG);
    return ok ? x[((size_t)b * IMG + r) * IMG + c] : 0.0f;
}

// Pack two f32 into one dword of f16x2 (v_cvt_pkrtz_f16_f32).
__device__ __forceinline__ unsigned pkh2(float a, float b) {
    fp16x2 p = __builtin_amdgcn_cvt_pkrtz(a, b);
    return __builtin_bit_cast(unsigned, p);
}

// Build the banded weight matrices for one scale (done once per block by wave 0).
// bh: 32x16 B-matrix for the horizontal pass (lane (n,half) holds col n, K rows e<8 ? 8*half+e : 8*half+e+8).
// av: 16x32 A-matrix for the vertical pass, scaled by K^2 (A layout K = 16*half+e).
template<int K>
__device__ __forceinline__ void build_weights(const float* __restrict__ kern,
                                              int n, int half,
                                              _Float16* __restrict__ bh_dst,
                                              _Float16* __restrict__ av_dst)
{
    constexpr int PADLO = (K - 1) / 2;      // TF-SAME: pad_low = (k-1)//2
    constexpr int SHIFT = SLAB - PADLO;
    // Separable 1D Gaussian = row sums of the normalized 2D kernel.
    float w[K];
    #pragma unroll
    for (int t = 0; t < K; ++t) {
        float s = 0.f;
        #pragma unroll
        for (int c = 0; c < K; ++c) s += kern[t * K + c];
        w[t] = s;
    }
    #pragma unroll
    for (int e = 0; e < 16; ++e) {
        int j  = (e < 8) ? (8 * half + e) : (8 * half + e + 8);  // B-layout K index
        int t  = j - n - SHIFT;
        float bval = 0.f;
        #pragma unroll
        for (int tt = 0; tt < K; ++tt) bval = (t == tt) ? w[tt] : bval;
        bh_dst[e] = (_Float16)bval;

        int j2 = 16 * half + e;             // A-layout K index
        int t2 = j2 - n - SHIFT;
        float aval = 0.f;
        #pragma unroll
        for (int tt = 0; tt < K; ++tt) aval = (t2 == tt) ? (w[tt] * (float)(K * K)) : aval;
        av_dst[e] = (_Float16)aval;
    }
}

// Plain (non-volatile) LDS load: lowers to ds_load_b128 x2.
__device__ __forceinline__ v16h lds_load_mat(const _Float16* p) {
    v8h lo = *(const v8h*)(p);
    v8h hi = *(const v8h*)(p + 8);
    return __builtin_shufflevector(lo, hi, 0,1,2,3,4,5,6,7,8,9,10,11,12,13,14,15);
}

__global__ __launch_bounds__(256) void densemap_kernel(
    const float* __restrict__ x,
    const float* __restrict__ k1, const float* __restrict__ k2,
    const float* __restrict__ k3, const float* __restrict__ k4,
    const float* __restrict__ k5, const float* __restrict__ k6,
    float* __restrict__ out)
{
    __shared__ __align__(16) _Float16 wmat[12][32][WPAD];

    const int lane = threadIdx.x & 31;
    const int wave = threadIdx.x >> 5;
    const int n    = lane & 15;
    const int half = lane >> 4;

    // Wave 0 builds all 12 banded weight matrices once; everyone else waits.
    if (wave == 0) {
        build_weights<1>(k1, n, half, &wmat[ 0][lane][0], &wmat[ 1][lane][0]);
        build_weights<2>(k2, n, half, &wmat[ 2][lane][0], &wmat[ 3][lane][0]);
        build_weights<3>(k3, n, half, &wmat[ 4][lane][0], &wmat[ 5][lane][0]);
        build_weights<4>(k4, n, half, &wmat[ 6][lane][0], &wmat[ 7][lane][0]);
        build_weights<5>(k5, n, half, &wmat[ 8][lane][0], &wmat[ 9][lane][0]);
        build_weights<6>(k6, n, half, &wmat[10][lane][0], &wmat[11][lane][0]);
    }
    __syncthreads();

    // Regression-slope coefficients c_k = (log2(k)-mean)/sum((log2-mean)^2).
    constexpr float ccoef[6] = { -0.34619504f, -0.12735791f, 0.00065366f,
                                  0.09147931f,  0.16192907f, 0.21949091f };

    #pragma unroll 1
    for (int it = 0; it < TPW; ++it) {
        const int tile = (blockIdx.x * WVS + wave) * TPW + it;   // 65536 tiles
        const int b  = tile >> 10;
        const int ty = (tile >> 5) & 31;
        const int tx = tile & 31;
        const int y0 = ty * 16, x0 = tx * 16;

        // ---- Load A0/A1: input slab rows [y0-4, y0+27], cols [x0-4, x0+27] as f16 ----
        v16h a0, a1;
        const bool interior = ((unsigned)(ty - 1) < 30u) & ((unsigned)(tx - 1) < 30u);
        if (interior) {
            // 16B-aligned vector loads (x0-4 is a multiple of 4 floats from a 16-multiple).
            const float* base0 = x + (((size_t)b * IMG + (y0 - SLAB + n)) * IMG + (x0 - SLAB));
            const float* base1 = base0 + (size_t)16 * IMG;
            f32x4 r0a = *(const f32x4*)(base0 +  8 * half);
            f32x4 r0b = *(const f32x4*)(base0 +  8 * half + 4);
            f32x4 r0c = *(const f32x4*)(base0 + 16 + 8 * half);
            f32x4 r0d = *(const f32x4*)(base0 + 20 + 8 * half);
            f32x4 r1a = *(const f32x4*)(base1 +  8 * half);
            f32x4 r1b = *(const f32x4*)(base1 +  8 * half + 4);
            f32x4 r1c = *(const f32x4*)(base1 + 16 + 8 * half);
            f32x4 r1d = *(const f32x4*)(base1 + 20 + 8 * half);
            #pragma unroll
            for (int q = 0; q < 4; ++q) {
                a0[q]      = (_Float16)r0a[q];  a0[4 + q]  = (_Float16)r0b[q];
                a0[8 + q]  = (_Float16)r0c[q];  a0[12 + q] = (_Float16)r0d[q];
                a1[q]      = (_Float16)r1a[q];  a1[4 + q]  = (_Float16)r1b[q];
                a1[8 + q]  = (_Float16)r1c[q];  a1[12 + q] = (_Float16)r1d[q];
            }
        } else {
            #pragma unroll
            for (int e = 0; e < 16; ++e) {
                int kidx = (e < 8) ? (8 * half + e) : (8 * half + e + 8);
                int gc   = x0 - SLAB + kidx;
                a0[e] = (_Float16)loadpix(x, b, y0 - SLAB + n,      gc);
                a1[e] = (_Float16)loadpix(x, b, y0 - SLAB + n + 16, gc);
            }
        }

        v8f acc = {};
        #pragma unroll
        for (int s = 0; s < 6; ++s) {
            // Opaque matrix index tied to the tile-loop counter: keeps the LDS
            // address inference (-> ds_load_b128) but prevents the compiler from
            // hoisting all 12 weight matrices into ~96 long-lived VGPRs.
            int mi = 2 * s;
            asm("" : "+v"(mi) : "v"(it));
            const v16h bh = lds_load_mat(&wmat[mi][lane][0]);
            const v16h av = lds_load_mat(&wmat[mi + 1][lane][0]);

            v8f cz = {};
            v8f h0 = __builtin_amdgcn_wmma_f32_16x16x32_f16(false, a0, false, bh, (short)0, cz, false, false);
            v8f h1 = __builtin_amdgcn_wmma_f32_16x16x32_f16(false, a1, false, bh, (short)0, cz, false, false);

            // Repack H (two f32 16x16 D tiles) into f16 B_v (32x16, K = slab row).
            // Pack to f16x2 first, exchange packed dwords across halves, then one
            // packed select per dword: 8 cvt_pk + 8 bpermute + 8 cndmask per scale.
            unsigned pk0[4], pk1[4], q0[4], q1[4];
            #pragma unroll
            for (int v = 0; v < 4; ++v) {
                pk0[v] = pkh2(h0[2 * v], h0[2 * v + 1]);   // D rows 8*half+2v,+1 of H0
                pk1[v] = pkh2(h1[2 * v], h1[2 * v + 1]);   // D rows 8*half+2v,+1 of H1
                q0[v]  = (unsigned)__shfl_xor((int)pk0[v], 16, 32);
                q1[v]  = (unsigned)__shfl_xor((int)pk1[v], 16, 32);
            }
            u32x8 bvp;
            #pragma unroll
            for (int v = 0; v < 4; ++v) {
                bvp[v]     = half ? q1[v]  : pk0[v];   // B K-rows 16*half + 2v,+1
                bvp[v + 4] = half ? pk1[v] : q0[v];    // B K-rows 16*half + 8 + 2v,+1
            }
            v16h bv = __builtin_bit_cast(v16h, bvp);

            v8f d = __builtin_amdgcn_wmma_f32_16x16x32_f16(false, av, false, bv, (short)0, cz, false, false);
            #pragma unroll
            for (int v = 0; v < 8; ++v)
                acc[v] += ccoef[s] * __log2f(fmaxf(d[v], 0.f) + 1.f);
        }

        // ---- Store D layout: lane (n,half) VGPR v -> row y0 + 8*half + v, col x0 + n ----
        float* o = out + ((size_t)b * IMG + y0) * IMG + x0;
        #pragma unroll
        for (int v = 0; v < 8; ++v) {
            int r = half * 8 + v;
            o[(size_t)r * IMG + n] = acc[v];
        }
    }
}

extern "C" void kernel_launch(void* const* d_in, const int* in_sizes, int n_in,
                              void* d_out, int out_size, void* d_ws, size_t ws_size,
                              hipStream_t stream) {
    const float* x  = (const float*)d_in[0];
    const float* k1 = (const float*)d_in[1];
    const float* k2 = (const float*)d_in[2];
    const float* k3 = (const float*)d_in[3];
    const float* k4 = (const float*)d_in[4];
    const float* k5 = (const float*)d_in[5];
    const float* k6 = (const float*)d_in[6];
    float* out = (float*)d_out;

    // 65536 wave-tiles / (8 waves * 8 tiles-per-wave) = 1024 blocks of 256 threads.
    densemap_kernel<<<1024, 256, 0, stream>>>(x, k1, k2, k3, k4, k5, k6, out);
}
